// SoftmaxMatcherBlock_50577534877890
// MI455X (gfx1250) — matrix operand order repaired and checked by reading, not compiled, not run on hardware
//
#include <hip/hip_runtime.h>
#include <hip/hip_bf16.h>

typedef _Float16 f16;
typedef __attribute__((ext_vector_type(8)))  float    v8f;
typedef __attribute__((ext_vector_type(8)))  _Float16 v8h;
typedef __attribute__((ext_vector_type(16))) _Float16 v16h;

static constexpr int Bn = 2, Cn = 64, Nn = 1024, Mn = 16384;
static constexpr int DV = 80;                 // value dims padded: 3 coords + 1 weight + 64 desc + 12 zero
static constexpr float TEMP = 0.02f;

__device__ __forceinline__ v16h cat8(v8h a, v8h b) {
  return __builtin_shufflevector(a, b, 0,1,2,3,4,5,6,7,8,9,10,11,12,13,14,15);
}
__device__ __forceinline__ v8f wmma_f16(v16h a, v16h b, v8f c) {
  return __builtin_amdgcn_wmma_f32_16x16x32_f16(false, a, false, b, (short)0, c, false, false);
}
__device__ __forceinline__ unsigned lds_off(const void* p) {
  return (unsigned)(uintptr_t)(__attribute__((address_space(3))) const void*)p;
}

// ---------------- Phase 0a: zn(src_desc) -> Qt, query-major f16 (B x N x C) ----------------
__global__ void zn_src_kernel(const float* __restrict__ src, f16* __restrict__ Qt) {
  int idx = blockIdx.x * blockDim.x + threadIdx.x;
  if (idx >= Bn * Nn) return;
  int b = idx / Nn, n = idx % Nn;
  const float* p = src + (size_t)b * Cn * Nn + n;
  float s = 0.f, s2 = 0.f;
  #pragma unroll 8
  for (int c = 0; c < Cn; ++c) { float x = p[(size_t)c * Nn]; s += x; s2 += x * x; }
  float mu  = s / Cn;
  float inv = rsqrtf((s2 - s * mu) / (Cn - 1));   // unbiased std (ddof=1)
  f16* q = Qt + ((size_t)b * Nn + n) * Cn;
  #pragma unroll 8
  for (int c = 0; c < Cn; ++c) q[c] = (f16)((p[(size_t)c * Nn] - mu) * inv);
}

// ---- Phase 0b: zn(tgt_desc) -> Kn key-major (B x M x C); build Vt dim-major (B x DV x M) ----
__global__ void zn_tgt_kernel(const float* __restrict__ tdesc, const float* __restrict__ tcoord,
                              const float* __restrict__ tw,
                              f16* __restrict__ Kn, f16* __restrict__ Vt) {
  int idx = blockIdx.x * blockDim.x + threadIdx.x;
  if (idx >= Bn * Mn) return;
  int b = idx / Mn, m = idx % Mn;
  const float* p = tdesc + (size_t)b * Cn * Mn + m;
  float s = 0.f, s2 = 0.f;
  #pragma unroll 8
  for (int c = 0; c < Cn; ++c) { float x = p[(size_t)c * Mn]; s += x; s2 += x * x; }
  float mu  = s / Cn;
  float inv = rsqrtf((s2 - s * mu) / (Cn - 1));
  f16* kn = Kn + ((size_t)b * Mn + m) * Cn;
  #pragma unroll 8
  for (int c = 0; c < Cn; ++c) kn[c] = (f16)((p[(size_t)c * Mn] - mu) * inv);
  f16* vt = Vt + (size_t)b * DV * Mn + m;
  #pragma unroll
  for (int i = 0; i < 3; ++i) vt[(size_t)i * Mn] = (f16)tcoord[((size_t)b * 3 + i) * Mn + m];
  vt[(size_t)3 * Mn] = (f16)tw[(size_t)b * Mn + m];
  #pragma unroll 8
  for (int c = 0; c < Cn; ++c) vt[(size_t)(4 + c) * Mn] = (f16)p[(size_t)c * Mn];
  #pragma unroll
  for (int d = 68; d < DV; ++d) vt[(size_t)d * Mn] = (f16)0.f;
}

// Issue one double-buffer batch: 6 async-to-LDS instructions per wave (ASYNCcnt += 6).
// K tile: 32 keys x 64 ch f16 = 4 KB, contiguous in key-major Kn -> 256 lanes x B128.
// V tile: 80 dims x 32 keys f16 = 5120 B, 64 B per row -> 5 x B32 per lane, uniform.
__device__ __forceinline__ void issue_tile_loads(const f16* __restrict__ KnB,
                                                 const f16* __restrict__ VtB,
                                                 int k0, f16* Kb, f16* Vb, int tid) {
  {
    const f16* src = KnB + (size_t)k0 * Cn + tid * 8;
    unsigned dst = lds_off(Kb + tid * 8);
    unsigned long long ga = (unsigned long long)(uintptr_t)src;
    asm volatile("global_load_async_to_lds_b128 %0, %1, off" :: "v"(dst), "v"(ga) : "memory");
  }
  #pragma unroll
  for (int j = 0; j < 5; ++j) {
    int o   = tid * 4 + j * 1024;          // byte offset within 5120 B tile
    int d   = o >> 6;                      // dim row (64 B per row)
    int rem = o & 63;
    const char* src = (const char*)VtB + ((size_t)d * Mn + k0) * 2 + rem;
    unsigned dst = lds_off((const char*)Vb + o);
    unsigned long long ga = (unsigned long long)(uintptr_t)src;
    asm volatile("global_load_async_to_lds_b32 %0, %1, off" :: "v"(dst), "v"(ga) : "memory");
  }
}

// ---------------- Phase 1: fused flash attention; 8 waves/block share K/V tiles ----------------
__launch_bounds__(256)
__global__ void attn_kernel(const f16* __restrict__ Qt, const f16* __restrict__ Kn,
                            const f16* __restrict__ Vt, float* __restrict__ Ows) {
  __shared__ f16 Kbuf[2][32 * Cn];          // 2 x 4 KB double buffer
  __shared__ f16 Vbuf[2][DV * 32];          // 2 x 5 KB double buffer
  __shared__ f16 Plds[8][32 * 16];          // per-wave P^T staging

  const int tid  = threadIdx.x;
  const int lane = tid & 31;
  const int wave = tid >> 5;
  const int tile = blockIdx.x * 8 + wave;   // 128 tiles; all 8 waves of a block share batch b
  const int b  = tile >> 6;                 // 64 tiles per batch
  const int n0 = (tile & 63) << 4;

  const f16* QtB = Qt + (size_t)b * Cn * Nn;
  const f16* KnB = Kn + (size_t)b * Mn * Cn;
  const f16* VtB = Vt + (size_t)b * DV * Mn;
  f16* P = Plds[wave];

  // B-fragment of Q^T (channels x queries): lane = query column; channels in table order.
  const f16* qp = QtB + (size_t)(n0 + (lane & 15)) * Cn + ((lane >> 4) << 3);
  v16h bq0 = cat8(*(const v8h*)qp,        *(const v8h*)(qp + 16));   // channels 0..31
  v16h bq1 = cat8(*(const v8h*)(qp + 32), *(const v8h*)(qp + 48));   // channels 32..63

  const int krow = lane & 15;       // A-matrix row within tile
  const int hi   = lane >> 4;       // lane half selects K-subblock
  const int ch   = hi << 3;         // 0 or 8
  const int q    = lane & 15;       // query this lane's softmax state covers

  v8f o[5];
  #pragma unroll
  for (int t = 0; t < 5; ++t) o[t] = (v8f){};
  float mrun = -1e30f, lrun = 0.f;
  const float e2 = 1.4426950408889634f / (Cn * TEMP);  // log2(e) * 1/(C*TEMP)

  int cur = 0;
  issue_tile_loads(KnB, VtB, 0, Kbuf[0], Vbuf[0], tid);

  for (int k0 = 0; k0 < Mn; k0 += 32) {
    // Prefetch next tile into the other buffer (wrap-around batch on last iter keeps
    // the ASYNCcnt discipline branch-free; it is never consumed).
    int k0n = (k0 + 32) & (Mn - 1);
    issue_tile_loads(KnB, VtB, k0n, Kbuf[cur ^ 1], Vbuf[cur ^ 1], tid);
    // 12 outstanding -> wait to 6: current buffer's batch has landed (in-order completion).
    asm volatile("s_wait_asynccnt 0x6" ::: "memory");
    __syncthreads();                                   // publish LDS across waves

    const f16* Kc = Kbuf[cur];
    const f16* Vc = Vbuf[cur];

    // ---- S^T = K_tile x Q^T for two 16-key subtiles (keys = M dim of WMMA) ----
    v8f dA, dB;
    {
      const f16* kp = Kc + (size_t)krow * Cn + ch;
      v16h A0 = cat8(*(const v8h*)kp,        *(const v8h*)(kp + 16));
      v16h A1 = cat8(*(const v8h*)(kp + 32), *(const v8h*)(kp + 48));
      v8f d = (v8f){};
      d = wmma_f16(A0, bq0, d);
      d = wmma_f16(A1, bq1, d);
      dA = d;
    }
    {
      const f16* kp = Kc + (size_t)(16 + krow) * Cn + ch;
      v16h A0 = cat8(*(const v8h*)kp,        *(const v8h*)(kp + 16));
      v16h A1 = cat8(*(const v8h*)(kp + 32), *(const v8h*)(kp + 48));
      v8f d = (v8f){};
      d = wmma_f16(A0, bq0, d);
      d = wmma_f16(A1, bq1, d);
      dB = d;
    }
    // ---- online softmax over keys (per-lane 16 values + cross-half combine) ----
    float tmax = -1e30f;
    #pragma unroll
    for (int i = 0; i < 8; ++i) tmax = fmaxf(tmax, fmaxf(dA[i], dB[i]));
    tmax = fmaxf(tmax, __shfl_xor(tmax, 16, 32));
    float mnew  = fmaxf(mrun, tmax);
    float alpha = __builtin_amdgcn_exp2f((mrun - mnew) * e2);
    float tsum  = 0.f;
    #pragma unroll
    for (int i = 0; i < 8; ++i) {
      float pa = __builtin_amdgcn_exp2f((dA[i] - mnew) * e2);
      float pb = __builtin_amdgcn_exp2f((dB[i] - mnew) * e2);
      tsum += pa + pb;
      int kk = i + ch;                       // this lane's key row in subtile
      P[q * 32 + kk]        = (f16)pa;
      P[q * 32 + kk + 16]   = (f16)pb;
    }
    tsum += __shfl_xor(tsum, 16, 32);
    lrun = lrun * alpha + tsum;
    mrun = mnew;
    #pragma unroll
    for (int t = 0; t < 5; ++t)
      #pragma unroll
      for (int i = 0; i < 8; ++i) o[t][i] *= alpha;
    // wave-local LDS store->load visibility (DS ops are in-order per wave)
    asm volatile("s_wait_dscnt 0x0" ::: "memory");
    // B-fragment of P^T (32 keys x 16 queries): lane = query column; keys in table order.
    v16h bp = cat8(*(const v8h*)(P + q * 32 + ch), *(const v8h*)(P + q * 32 + ch + 16));
    // ---- O^T += V^T x P^T over 5 dim-tiles ----
    #pragma unroll
    for (int t = 0; t < 5; ++t) {
      const f16* vp = Vc + (size_t)(t * 16 + krow) * 32 + ch;
      v16h Av = cat8(*(const v8h*)vp, *(const v8h*)(vp + 16));
      o[t] = wmma_f16(Av, bp, o[t]);
    }
    __syncthreads();                       // all waves done reading before buffer reuse
    cur ^= 1;
  }

  // ---- store O^T / l  ->  Ows (B x DV x N), D-layout: (d,q) = vgpr d%8, lane q+16*(d/8)
  float invl = 1.0f / lrun;
  #pragma unroll
  for (int t = 0; t < 5; ++t)
    #pragma unroll
    for (int i = 0; i < 8; ++i) {
      int d = t * 16 + i + ch;
      Ows[((size_t)b * DV + d) * Nn + n0 + q] = o[t][i] * invl;
    }
}

// ---------------- Phase 2: split outputs; final zn over desc dims ----------------
__global__ void finalize_kernel(const float* __restrict__ Ows, float* __restrict__ out) {
  int idx = blockIdx.x * blockDim.x + threadIdx.x;
  if (idx >= Bn * Nn) return;
  int b = idx / Nn, n = idx % Nn;
  const float* o = Ows + (size_t)b * DV * Nn + n;
  // pseudo_coords: (B,3,N) at offset 0
  #pragma unroll
  for (int i = 0; i < 3; ++i)
    out[((size_t)b * 3 + i) * Nn + n] = o[(size_t)i * Nn];
  // pseudo_weights: (B,1,N) at offset 6144
  out[6144 + (size_t)b * Nn + n] = o[(size_t)3 * Nn];
  // pseudo_descs: zn over C, (B,64,N) at offset 8192
  float s = 0.f, s2 = 0.f;
  #pragma unroll 8
  for (int c = 0; c < Cn; ++c) { float x = o[(size_t)(4 + c) * Nn]; s += x; s2 += x * x; }
  float mu  = s / Cn;
  float inv = rsqrtf((s2 - s * mu) / (Cn - 1));
  #pragma unroll 8
  for (int c = 0; c < Cn; ++c)
    out[8192 + ((size_t)b * Cn + c) * Nn + n] = (o[(size_t)(4 + c) * Nn] - mu) * inv;
}

extern "C" void kernel_launch(void* const* d_in, const int* in_sizes, int n_in,
                              void* d_out, int out_size, void* d_ws, size_t ws_size,
                              hipStream_t stream) {
  // inputs: 0 src_coords (unused), 1 tgt_coords, 2 tgt_weights, 3 src_desc, 4 tgt_desc
  const float* tgt_coords  = (const float*)d_in[1];
  const float* tgt_weights = (const float*)d_in[2];
  const float* src_desc    = (const float*)d_in[3];
  const float* tgt_desc    = (const float*)d_in[4];
  char* ws = (char*)d_ws;
  f16*   Qt  = (f16*)(ws);                                  // 2*64*1024*2   = 256 KB
  f16*   Kn  = (f16*)(ws + (size_t)262144);                 // 2*16384*64*2  = 4 MB
  f16*   Vt  = (f16*)(ws + (size_t)4456448);                // 2*80*16384*2  = 5 MB
  float* Ows = (float*)(ws + (size_t)9699328);              // 2*80*1024*4   = 640 KB
  float* out = (float*)d_out;

  zn_src_kernel<<<(Bn * Nn + 255) / 256, 256, 0, stream>>>(src_desc, Qt);
  zn_tgt_kernel<<<(Bn * Mn + 255) / 256, 256, 0, stream>>>(tgt_desc, tgt_coords, tgt_weights, Kn, Vt);
  attn_kernel<<<16, 256, 0, stream>>>(Qt, Kn, Vt, Ows);
  finalize_kernel<<<(Bn * Nn + 255) / 256, 256, 0, stream>>>(Ows, out);
}
